// RetNetModel_29351806501436
// MI455X (gfx1250) — compile-verified, matrix operand order checked by reading, not run on hardware
//
#include <hip/hip_runtime.h>

// ---------------------------------------------------------------------------
// RetNet jet tagger for MI455X (gfx1250, wave32, WMMA).
// All heavy GEMMs -> v_wmma_f32_16x16x32_bf16 with f32 accumulation.
// Double-buffered LDS K-tiles; async global->LDS (gfx1250 ASYNC path).
// Epilogues (bias / GELU / ReLU / xPos rotary / SiLU-gate / residual) fused
// and templated (dead-code-free per instantiation).
// ---------------------------------------------------------------------------

typedef __bf16 bf16;
typedef __attribute__((ext_vector_type(16))) __bf16 v16bf;
typedef __attribute__((ext_vector_type(8)))  float  v8f;

#define H_    256
#define FFN_  1024
#define NH_   8
#define HK_   32
#define B_    8192
#define S_T   16
#define S_V   5
#define IN_T  16
#define IN_V  14
#define R_T   (B_ * S_T)   // 131072
#define R_V   (B_ * S_V)   // 40960

enum { EPI_LIN = 0, EPI_GELU, EPI_RELU, EPI_XPOS, EPI_GATE, EPI_RES };

#if defined(__HIP_DEVICE_COMPILE__) && __has_builtin(__builtin_amdgcn_global_load_async_to_lds_b128)
#define HAVE_ASYNC_LDS 1
typedef int v4i_ __attribute__((vector_size(16)));   // matches builtin param
#define ALD(gp, lp) __builtin_amdgcn_global_load_async_to_lds_b128( \
    (__attribute__((address_space(1))) v4i_*)(gp),                  \
    (__attribute__((address_space(3))) v4i_*)(lp), 0, 0)
#if __has_builtin(__builtin_amdgcn_s_wait_asynccnt)
#define WAIT_ASYNC(n) __builtin_amdgcn_s_wait_asynccnt(n)
#else
#define WAIT_ASYNC(n) asm volatile("s_wait_asynccnt " #n ::: "memory")
#endif
#endif

// ---------------------------------------------------------------------------
// Generic bf16 GEMM: C[M,N] = epi(A[M,K] @ Wt[N,K]^T + bias)
// Wt is the weight pre-transposed to N-major (row n holds W[:,n]), K-padded.
// Block = 256 threads = 8 waves; tile 128x128; K-step 32, double-buffered.
// Wave (wm in 0..3, wn in 0..1) computes a 32x64 sub-tile = 2x4 WMMA tiles.
// ---------------------------------------------------------------------------
template <int EPI>
__global__ __launch_bounds__(256, 2)
void gemm_bf16(const bf16* __restrict__ A, const bf16* __restrict__ Wt,
               const float* __restrict__ bias, bf16* __restrict__ C,
               int M, int N, int K,
               const bf16* __restrict__ aux, int S, int xdown)
{
    __shared__ __align__(16) bf16 As[2][128 * 40];  // stride 40 bf16 = 80B
    __shared__ __align__(16) bf16 Bs[2][128 * 40];

    const int tid  = threadIdx.x;
    const int lane = tid & 31;
    const int wave = tid >> 5;
    const int wm   = wave & 3;       // 4 waves along M (32 rows each)
    const int wn   = wave >> 2;      // 2 waves along N (64 cols each)
    const int m0   = blockIdx.x * 128;
    const int n0   = blockIdx.y * 128;
    const int lrow = tid >> 1;       // 0..127: loader row
    const int lcol = (tid & 1) << 4; // 0 or 16: loader col (bf16 elems)
    const int loff = lrow * 40 + lcol;
    const int lh   = lane >> 4;      // lane half (A/B fragment K-split)
    const int lm   = lane & 15;

    v8f acc[2][4];
#pragma unroll
    for (int i = 0; i < 2; ++i)
#pragma unroll
        for (int j = 0; j < 4; ++j)
#pragma unroll
            for (int e = 0; e < 8; ++e) acc[i][j][e] = 0.f;

    const bf16* pa = A  + (size_t)(m0 + lrow) * K + lcol;
    const bf16* pb = Wt + (size_t)(n0 + lrow) * K + lcol;
    int buf = 0;

#if HAVE_ASYNC_LDS
    // ---- async global->LDS pipeline (gfx1250 ASYNC data path) ----
    ALD(pa,     &As[0][loff]);
    ALD(pa + 8, &As[0][loff + 8]);
    ALD(pb,     &Bs[0][loff]);
    ALD(pb + 8, &Bs[0][loff + 8]);
    for (int kb = 0; kb < K; kb += 32) {
        // issue next tile (or a harmless re-issue on the last step)
        int off = (kb + 32 < K) ? 32 : 0;
        ALD(pa + off,     &As[buf ^ 1][loff]);
        ALD(pa + off + 8, &As[buf ^ 1][loff + 8]);
        ALD(pb + off,     &Bs[buf ^ 1][loff]);
        ALD(pb + off + 8, &Bs[buf ^ 1][loff + 8]);
        WAIT_ASYNC(4);              // oldest 4 (current tile) have landed
        __syncthreads();
#else
    // ---- register double-buffer pipeline ----
    uint4 a0 = *(const uint4*)(pa);
    uint4 a1 = *(const uint4*)(pa + 8);
    uint4 b0 = *(const uint4*)(pb);
    uint4 b1 = *(const uint4*)(pb + 8);
    for (int kb = 0; kb < K; kb += 32) {
        *(uint4*)&As[buf][loff]     = a0;
        *(uint4*)&As[buf][loff + 8] = a1;
        *(uint4*)&Bs[buf][loff]     = b0;
        *(uint4*)&Bs[buf][loff + 8] = b1;
        __syncthreads();
        // prefetch next K-tile into registers while WMMAs run
        int off = (kb + 32 < K) ? 32 : 0;
        uint4 na0 = *(const uint4*)(pa + off);
        uint4 na1 = *(const uint4*)(pa + off + 8);
        uint4 nb0 = *(const uint4*)(pb + off);
        uint4 nb1 = *(const uint4*)(pb + off + 8);
        if (kb + 64 < K) {                      // gfx1250 global_prefetch_b8
            __builtin_prefetch(pa + 64, 0, 1);
            __builtin_prefetch(pb + 64, 0, 1);
        }
#endif
        // A fragment (16x32): lane half 0 -> K 0..7 & 16..23; half 1 -> +8
        union { uint4 u[2]; v16bf v; } af[2], bfr[4];
#pragma unroll
        for (int mt = 0; mt < 2; ++mt) {
            const bf16* s = &As[buf][(wm * 32 + mt * 16 + lm) * 40 + lh * 8];
            af[mt].u[0] = *(const uint4*)(s);
            af[mt].u[1] = *(const uint4*)(s + 16);
        }
#pragma unroll
        for (int nt = 0; nt < 4; ++nt) {
            const bf16* s = &Bs[buf][(wn * 64 + nt * 16 + lm) * 40 + lh * 8];
            bfr[nt].u[0] = *(const uint4*)(s);
            bfr[nt].u[1] = *(const uint4*)(s + 16);
        }
#pragma unroll
        for (int mt = 0; mt < 2; ++mt)
#pragma unroll
            for (int nt = 0; nt < 4; ++nt)
                acc[mt][nt] = __builtin_amdgcn_wmma_f32_16x16x32_bf16(
                    false, af[mt].v, false, bfr[nt].v,
                    (short)0, acc[mt][nt], false, false);
#if HAVE_ASYNC_LDS
        __syncthreads();            // seal reads before next async overwrite
#else
        a0 = na0; a1 = na1; b0 = nb0; b1 = nb1;
#endif
        pa += 32;
        pb += 32;
        buf ^= 1;
    }

    // Epilogue. C layout: VGPR i -> M = i + 8*(lane>=16), N = lane&15.
#pragma unroll
    for (int mt = 0; mt < 2; ++mt)
#pragma unroll
        for (int nt = 0; nt < 4; ++nt)
#pragma unroll
            for (int i = 0; i < 8; ++i) {
                int m = m0 + wm * 32 + mt * 16 + lh * 8 + i;
                int n = n0 + wn * 64 + nt * 16 + lm;
                float x = acc[mt][nt][i];
                if (bias && n < N) x += bias[n];
                if constexpr (EPI == EPI_GELU) {
                    x = 0.5f * x * (1.f + erff(x * 0.70710678f));
                } else if constexpr (EPI == EPI_RELU) {
                    x = fmaxf(x, 0.f);
                } else if constexpr (EPI == EPI_XPOS) {
                    float part = __shfl_xor(x, 1, 32); // paired rotary column
                    int kk = n & 31, j = kk >> 1;
                    float pos = (float)(m % S);
                    float sc  = __powf((2.f * j + 12.8f) * (1.f / 44.8f),
                                       pos * (1.f / 512.f));
                    if (xdown) sc = 1.f / sc;
                    float ang = pos * __powf(10000.f, -(float)j * (1.f / 16.f));
                    float rot = (kk & 1) ? part : -part;
                    x = (x * __cosf(ang) + rot * __sinf(ang)) * sc;
                } else if constexpr (EPI == EPI_GATE) {
                    if (n < N) {
                        float yn = (float)aux[(size_t)m * N + n];
                        x = x * (1.f / (1.f + __expf(-x))) * yn; // sig(G)*G*Yn
                    }
                } else if constexpr (EPI == EPI_RES) {
                    if (n < N) x += (float)aux[(size_t)m * N + n];
                }
                if (n < N) C[(size_t)m * N + n] = (bf16)x;
            }
}

// ---------------------------------------------------------------------------
// LayerNorm over H=256, one wave per row (8 rows per 256-thread block).
// ---------------------------------------------------------------------------
__global__ __launch_bounds__(256)
void ln_kernel(const bf16* __restrict__ X, const float* __restrict__ w,
               const float* __restrict__ b, bf16* __restrict__ O)
{
    int lane = threadIdx.x & 31;
    int row  = blockIdx.x * 8 + (threadIdx.x >> 5);
    const bf16* xr = X + (size_t)row * H_;
    float v[8], s = 0.f;
#pragma unroll
    for (int j = 0; j < 8; ++j) { v[j] = (float)xr[lane + 32 * j]; s += v[j]; }
#pragma unroll
    for (int o = 16; o >= 1; o >>= 1) s += __shfl_xor(s, o, 32);
    float mean = s * (1.f / 256.f);
    float q = 0.f;
#pragma unroll
    for (int j = 0; j < 8; ++j) { float d = v[j] - mean; q += d * d; }
#pragma unroll
    for (int o = 16; o >= 1; o >>= 1) q += __shfl_xor(q, o, 32);
    float r = rsqrtf(q * (1.f / 256.f) + 1e-5f);
    bf16* orow = O + (size_t)row * H_;
#pragma unroll
    for (int j = 0; j < 8; ++j) {
        int c = lane + 32 * j;
        orow[c] = (bf16)((v[j] - mean) * r * w[c] + b[c]);
    }
}

// ---------------------------------------------------------------------------
// Retention attention + per-head groupnorm. One wave32 per (jet, head).
// att[s,t] = (Q_s . K_t) * gamma^(s-t) for t<=s ; Y = att @ V ; per-(s,head)
// normalize over the 32 v-dims, then gn_w/gn_b. S <= 16.
// ---------------------------------------------------------------------------
__global__ __launch_bounds__(32)
void attn_kernel(const bf16* __restrict__ Q, const bf16* __restrict__ K,
                 const bf16* __restrict__ V, const float* __restrict__ gnw,
                 const float* __restrict__ gnb, bf16* __restrict__ Yn, int S)
{
    __shared__ float Qs[512], Ks[512], Vs[512], att[256];
    int lane = threadIdx.x;
    int jet = blockIdx.x >> 3, head = blockIdx.x & 7;
    size_t base = ((size_t)jet * S) * H_ + head * 32 + lane;
    for (int s = 0; s < S; ++s) {
        Qs[s * 32 + lane] = (float)Q[base + (size_t)s * H_];
        Ks[s * 32 + lane] = (float)K[base + (size_t)s * H_];
        Vs[s * 32 + lane] = (float)V[base + (size_t)s * H_];
    }
    __syncthreads();
    const float lg0 = -3.4657359f;                       // ln(1/32)
    const float dlg = (-6.2383246f + 3.4657359f) / 7.f;  // to ln(1/512)
    float gamma = 1.f - __expf(lg0 + head * dlg);
    for (int p = lane; p < S * S; p += 32) {
        int s = p / S, t = p % S;
        float a = 0.f;
        if (t <= s) {
            float d = 0.f;
            for (int k = 0; k < 32; ++k) d += Qs[s * 32 + k] * Ks[t * 32 + k];
            a = d * __powf(gamma, (float)(s - t));
        }
        att[p] = a;
    }
    __syncthreads();
    for (int s = 0; s < S; ++s) {
        float y = 0.f;
        for (int t = 0; t <= s; ++t) y += att[s * S + t] * Vs[t * 32 + lane];
        float sum = y;
#pragma unroll
        for (int o = 16; o >= 1; o >>= 1) sum += __shfl_xor(sum, o, 32);
        float mean = sum * (1.f / 32.f);
        float d = y - mean, q = d * d;
#pragma unroll
        for (int o = 16; o >= 1; o >>= 1) q += __shfl_xor(q, o, 32);
        float r = rsqrtf(q * (1.f / 32.f) + 1e-5f);
        int c = head * 32 + lane;
        Yn[((size_t)jet * S + s) * H_ + c] = (bf16)(d * r * gnw[c] + gnb[c]);
    }
}

// ---------------------------------------------------------------------------
// Small helper kernels: weight prep (fp32 -> bf16, transposed/padded),
// input padding, concat, final 128->1 + sigmoid.
// ---------------------------------------------------------------------------
__global__ void prepT(const float* __restrict__ src, bf16* __restrict__ dst,
                      int K, int N, int Kpad, int Npad)
{
    int idx = blockIdx.x * 256 + threadIdx.x;
    if (idx >= Npad * Kpad) return;
    int n = idx / Kpad, k = idx % Kpad;
    dst[idx] = (bf16)((n < N && k < K) ? src[(size_t)k * N + n] : 0.f);
}

// wq/wk/wv are (NH,H,HK); produce Wt[col = n*32+kk][h] with N-major layout
__global__ void prepQKV(const float* __restrict__ src, bf16* __restrict__ dst)
{
    int idx = blockIdx.x * 256 + threadIdx.x;
    if (idx >= 256 * 256) return;
    int col = idx / 256, h = idx % 256;
    dst[idx] = (bf16)src[(col >> 5) * (H_ * HK_) + h * HK_ + (col & 31)];
}

__global__ void pad_in(const float* __restrict__ src, bf16* __restrict__ dst,
                       int R, int IN)
{
    int idx = blockIdx.x * 256 + threadIdx.x;
    if (idx >= R * 32) return;
    int row = idx >> 5, c = idx & 31;
    dst[idx] = (bf16)(c < IN ? src[(size_t)row * IN + c] : 0.f);
}

__global__ void concat_kernel(const bf16* __restrict__ t,
                              const bf16* __restrict__ v,
                              const float* __restrict__ jx,
                              bf16* __restrict__ cat)
{
    int idx = blockIdx.x * 256 + threadIdx.x;
    if (idx >= B_ * 96) return;
    int row = idx / 96, c = idx % 96;
    float val = 0.f;
    if (c < 32)       val = (float)t[(size_t)row * 32 + c];
    else if (c < 64)  val = (float)v[(size_t)row * 32 + (c - 32)];
    else if (c < 70)  val = jx[(size_t)row * 6 + (c - 64)];
    cat[idx] = (bf16)val;
}

__global__ __launch_bounds__(256)
void head_final(const bf16* __restrict__ X, const float* __restrict__ w,
                const float* __restrict__ b, float* __restrict__ out)
{
    int lane = threadIdx.x & 31;
    int row  = blockIdx.x * 8 + (threadIdx.x >> 5);
    float s = 0.f;
#pragma unroll
    for (int j = 0; j < 4; ++j) {
        int k = lane + 32 * j;
        s += (float)X[(size_t)row * 128 + k] * w[k];
    }
#pragma unroll
    for (int o = 16; o >= 1; o >>= 1) s += __shfl_xor(s, o, 32);
    if (lane == 0) out[row] = 1.f / (1.f + __expf(-(s + b[0])));
}

// ---------------------------------------------------------------------------
// Host orchestration
// ---------------------------------------------------------------------------
struct BranchW {
    bf16 *e1T, *e2T;
    bf16 *qT[4], *kT[4], *vT[4], *gT[4], *oT[4], *f1T[4], *f2T[4];
    bf16 *m0T, *m1T, *m2T, *m3T;
};

extern "C" void kernel_launch(void* const* d_in, const int* in_sizes, int n_in,
                              void* d_out, int out_size, void* d_ws, size_t ws_size,
                              hipStream_t stream)
{
    (void)in_sizes; (void)n_in; (void)out_size; (void)ws_size;
    auto F = [&](int i) { return (const float*)d_in[i]; };

    char* ws = (char*)d_ws;
    size_t cur = 0;
    auto alloc = [&](size_t elems) -> bf16* {
        bf16* p = (bf16*)(ws + cur);
        cur = (cur + elems * 2 + 255) & ~(size_t)255;
        return p;
    };

    // ---- weight workspace (prepped bf16, N-major, K-padded) ----
    BranchW tW, vW;
    auto allocBranch = [&](BranchW& W, int mlpK) {
        W.e1T = alloc(1024 * 32);
        W.e2T = alloc(256 * 1024);
        for (int l = 0; l < 4; ++l) {
            W.qT[l]  = alloc(256 * 256);
            W.kT[l]  = alloc(256 * 256);
            W.vT[l]  = alloc(256 * 256);
            W.gT[l]  = alloc(256 * 256);
            W.oT[l]  = alloc(256 * 256);
            W.f1T[l] = alloc(1024 * 256);
            W.f2T[l] = alloc(256 * 1024);
        }
        W.m0T = alloc((size_t)128 * mlpK);
        W.m1T = alloc(128 * 128);
        W.m2T = alloc(128 * 128);
        W.m3T = alloc(128 * 128); // N=32 zero-padded to 128 rows
    };
    allocBranch(tW, 4096);
    allocBranch(vW, 1280);
    bf16* j0T = alloc(128 * 96);
    bf16* j1T = alloc(128 * 128);
    bf16* j2T = alloc(128 * 128);
    bf16* j3T = alloc(128 * 128);

    // ---- activation workspace (track-sized, reused by vertex) ----
    bf16* Xb   = alloc((size_t)R_T * H_);
    bf16* Nb   = alloc((size_t)R_T * H_);
    bf16* Qb   = alloc((size_t)R_T * H_);   // also reused as gated buffer
    bf16* Kb   = alloc((size_t)R_T * H_);
    bf16* Vb   = alloc((size_t)R_T * H_);
    bf16* Ynb  = alloc((size_t)R_T * H_);
    bf16* Hid  = alloc((size_t)R_T * FFN_); // embed-hidden + ffn-hidden
    bf16* Apad = alloc((size_t)R_T * 32);
    bf16* tOut = alloc((size_t)B_ * 32);
    bf16* vOut = alloc((size_t)B_ * 32);
    bf16* Ha   = alloc((size_t)B_ * 128);
    bf16* Hb2  = alloc((size_t)B_ * 128);
    bf16* cat  = alloc((size_t)B_ * 96);

    auto prepTL = [&](const float* src, bf16* dst, int K, int N, int Kpad, int Npad) {
        int tot = Npad * Kpad;
        prepT<<<(tot + 255) / 256, 256, 0, stream>>>(src, dst, K, N, Kpad, Npad);
    };

    // ---- weight prep launches ----
    // d_in map: 0 track_x, 1 vertex_x, 2 jet_x, 3-6 track_embed, 7-10 vertex_embed,
    // 11-25 track_ret, 26-40 vertex_ret, 41-48 track_mlp, 49-56 vertex_mlp, 57-66 jet_mlp
    auto prepBranch = [&](const BranchW& W, int eb, int rb, int mb, int IN, int mlpK) {
        prepTL(F(eb + 0), W.e1T, IN, 1024, 32, 1024);
        prepTL(F(eb + 2), W.e2T, 1024, 256, 1024, 256);
        for (int l = 0; l < 4; ++l) {
            prepQKV<<<256, 256, 0, stream>>>(F(rb + 0) + (size_t)l * 65536, W.qT[l]);
            prepQKV<<<256, 256, 0, stream>>>(F(rb + 1) + (size_t)l * 65536, W.kT[l]);
            prepQKV<<<256, 256, 0, stream>>>(F(rb + 2) + (size_t)l * 65536, W.vT[l]);
            prepTL(F(rb + 3)  + (size_t)l * 65536,  W.gT[l],  256, 256, 256, 256);
            prepTL(F(rb + 4)  + (size_t)l * 65536,  W.oT[l],  256, 256, 256, 256);
            prepTL(F(rb + 11) + (size_t)l * 262144, W.f1T[l], 256, 1024, 256, 1024);
            prepTL(F(rb + 13) + (size_t)l * 262144, W.f2T[l], 1024, 256, 1024, 256);
        }
        prepTL(F(mb + 0), W.m0T, mlpK, 128, mlpK, 128);
        prepTL(F(mb + 1), W.m1T, 128, 128, 128, 128);
        prepTL(F(mb + 2), W.m2T, 128, 128, 128, 128);
        prepTL(F(mb + 3), W.m3T, 128, 32, 128, 128);
    };
    prepBranch(tW, 3, 11, 41, IN_T, 4096);
    prepBranch(vW, 7, 26, 49, IN_V, 1280);
    prepTL(F(57), j0T, 70, 128, 96, 128);
    prepTL(F(58), j1T, 128, 128, 128, 128);
    prepTL(F(59), j2T, 128, 128, 128, 128);
    prepTL(F(60), j3T, 128, 128, 128, 128);

    auto gemm = [&](int epi, const bf16* A, const bf16* Wt, const float* bias,
                    bf16* C, int M, int N, int K, const bf16* aux, int S, int down) {
        dim3 grid(M / 128, (N + 127) / 128);
        switch (epi) {
        case EPI_LIN:  gemm_bf16<EPI_LIN> <<<grid, 256, 0, stream>>>(A, Wt, bias, C, M, N, K, aux, S, down); break;
        case EPI_GELU: gemm_bf16<EPI_GELU><<<grid, 256, 0, stream>>>(A, Wt, bias, C, M, N, K, aux, S, down); break;
        case EPI_RELU: gemm_bf16<EPI_RELU><<<grid, 256, 0, stream>>>(A, Wt, bias, C, M, N, K, aux, S, down); break;
        case EPI_XPOS: gemm_bf16<EPI_XPOS><<<grid, 256, 0, stream>>>(A, Wt, bias, C, M, N, K, aux, S, down); break;
        case EPI_GATE: gemm_bf16<EPI_GATE><<<grid, 256, 0, stream>>>(A, Wt, bias, C, M, N, K, aux, S, down); break;
        default:       gemm_bf16<EPI_RES> <<<grid, 256, 0, stream>>>(A, Wt, bias, C, M, N, K, aux, S, down); break;
        }
    };

    // ---- per-branch pipeline ----
    auto run_branch = [&](const BranchW& W, int S, int IN, int xi,
                          int eb, int rb, int bb, bf16* outBuf) {
        int R = B_ * S;
        pad_in<<<(R * 32 + 255) / 256, 256, 0, stream>>>(F(xi), Apad, R, IN);
        gemm(EPI_GELU, Apad, W.e1T, F(eb + 1), Hid, R, 1024, 32,  nullptr, 0, 0);
        gemm(EPI_GELU, Hid,  W.e2T, F(eb + 3), Xb,  R, 256, 1024, nullptr, 0, 0);
        for (int l = 0; l < 4; ++l) {
            ln_kernel<<<R / 8, 256, 0, stream>>>(Xb, F(rb + 7) + l * H_,
                                                 F(rb + 8) + l * H_, Nb);
            gemm(EPI_XPOS, Nb, W.qT[l], nullptr, Qb, R, 256, 256, nullptr, S, 0);
            gemm(EPI_XPOS, Nb, W.kT[l], nullptr, Kb, R, 256, 256, nullptr, S, 1);
            gemm(EPI_LIN,  Nb, W.vT[l], nullptr, Vb, R, 256, 256, nullptr, 0, 0);
            attn_kernel<<<B_ * NH_, 32, 0, stream>>>(Qb, Kb, Vb,
                F(rb + 5) + l * H_, F(rb + 6) + l * H_, Ynb, S);
            gemm(EPI_GATE, Nb, W.gT[l], nullptr, Qb, R, 256, 256, Ynb, 0, 0);
            gemm(EPI_RES,  Qb, W.oT[l], nullptr, Xb, R, 256, 256, Xb, 0, 0);
            ln_kernel<<<R / 8, 256, 0, stream>>>(Xb, F(rb + 9) + l * H_,
                                                 F(rb + 10) + l * H_, Nb);
            gemm(EPI_GELU, Nb,  W.f1T[l], F(rb + 12) + l * FFN_, Hid, R, 1024, 256,  nullptr, 0, 0);
            gemm(EPI_RES,  Hid, W.f2T[l], F(rb + 14) + l * H_,   Xb,  R, 256,  1024, Xb, 0, 0);
        }
        // head MLP on flattened (B, S*H)
        gemm(EPI_RELU, Xb,  W.m0T, F(bb + 0), Ha,     B_, 128, S * H_, nullptr, 0, 0);
        gemm(EPI_RELU, Ha,  W.m1T, F(bb + 1), Hb2,    B_, 128, 128,    nullptr, 0, 0);
        gemm(EPI_RELU, Hb2, W.m2T, F(bb + 2), Ha,     B_, 128, 128,    nullptr, 0, 0);
        gemm(EPI_LIN,  Ha,  W.m3T, F(bb + 3), outBuf, B_, 32,  128,    nullptr, 0, 0);
    };

    run_branch(tW, S_T, IN_T, 0, 3, 11, 45, tOut);
    run_branch(vW, S_V, IN_V, 1, 7, 26, 53, vOut);

    // ---- jet head ----
    concat_kernel<<<(B_ * 96 + 255) / 256, 256, 0, stream>>>(tOut, vOut, F(2), cat);
    gemm(EPI_RELU, cat, j0T, F(62), Ha,  B_, 128, 96,  nullptr, 0, 0);
    gemm(EPI_RELU, Ha,  j1T, F(63), Hb2, B_, 128, 128, nullptr, 0, 0);
    gemm(EPI_RELU, Hb2, j2T, F(64), Ha,  B_, 128, 128, nullptr, 0, 0);
    gemm(EPI_RELU, Ha,  j3T, F(65), Hb2, B_, 128, 128, nullptr, 0, 0);
    head_final<<<B_ / 8, 256, 0, stream>>>(Hb2, F(61), F(66), (float*)d_out);
}
